// RiemannianSpaceAttention_8615704396366
// MI455X (gfx1250) — matrix operand order repaired and checked by reading, not compiled, not run on hardware
//
#include <hip/hip_runtime.h>

#define EPSV 1e-5f

typedef float v2f  __attribute__((ext_vector_type(2)));
typedef float v8f  __attribute__((ext_vector_type(8)));
typedef __bf16 v16bf __attribute__((ext_vector_type(16)));
typedef unsigned u32x4 __attribute__((ext_vector_type(4)));
typedef int i32x8 __attribute__((ext_vector_type(8)));
typedef int i32x4 __attribute__((ext_vector_type(4)));

// ---------- bf16 helpers (bit-level) ----------
__device__ inline __bf16 f2bf(float f) {
    unsigned u; __builtin_memcpy(&u, &f, 4);
    unsigned short h = (unsigned short)(u >> 16);   // truncate-to-bf16
    __bf16 r; __builtin_memcpy(&r, &h, 2);
    return r;
}

// ---------- WMMA fragment loaders ----------
__device__ inline v16bf ldA_f32(const float* A, int lda, int r0, int k0, int g, int m) {
    v16bf a;
    const float* row = A + (r0 + m) * lda;
#pragma unroll
    for (int e = 0; e < 16; ++e) {
        int kk = k0 + (g << 4) + e;
        a[e] = f2bf(row[kk]);
    }
    return a;
}
__device__ inline v16bf ldA_bf16(const __bf16* A, int lda, int r0, int k0, int g, int m) {
    v16bf a;
    const __bf16* row = A + (r0 + m) * lda;
#pragma unroll
    for (int e = 0; e < 16; ++e) {
        int kk = k0 + (g << 4) + e;
        a[e] = row[kk];
    }
    return a;
}
__device__ inline v16bf ldB_f32(const float* B, int ldb, int k0, int c0, int g, int m) {
    v16bf b;
#pragma unroll
    for (int e = 0; e < 16; ++e)
        b[e] = f2bf(B[(k0 + (g << 4) + e) * ldb + c0 + m]);
    return b;
}
__device__ inline v16bf ldB_bf16(const __bf16* B, int ldb, int k0, int c0, int g, int m) {
    v16bf b;
#pragma unroll
    for (int e = 0; e < 16; ++e)
        b[e] = B[(k0 + (g << 4) + e) * ldb + c0 + m];
    return b;
}
__device__ inline v16bf ldBT_bf16(const __bf16* B, int ldb, int k0, int c0, int g, int m) {
    v16bf b;
    const __bf16* row = B + (c0 + m) * ldb;
#pragma unroll
    for (int e = 0; e < 16; ++e)
        b[e] = row[k0 + (g << 4) + e];
    return b;
}

// ---------- 64x64 f32 matmul on v_wmma_f32_16x16x4_f32 ----------
__device__ inline void mm64_f32(const float* __restrict__ A, int lda,
                                const float* __restrict__ B, int ldb, bool BT,
                                float* __restrict__ C, int ldc,
                                int K, int wave, int g, int m) {
    for (int t = wave; t < 16; t += 8) {
        const int r0 = (t >> 2) << 4;
        const int c0 = (t & 3) << 4;
        v8f acc = {0.f, 0.f, 0.f, 0.f, 0.f, 0.f, 0.f, 0.f};
        for (int k0 = 0; k0 < K; k0 += 4) {
            v2f a, b;
            a.x = A[(r0 + m) * lda + (k0 + 2 * g)];
            a.y = A[(r0 + m) * lda + (k0 + 2 * g + 1)];
            if (BT) {
                b.x = B[(c0 + m) * ldb + (k0 + 2 * g)];
                b.y = B[(c0 + m) * ldb + (k0 + 2 * g + 1)];
            } else {
                b.x = B[(k0 + 2 * g) * ldb + (c0 + m)];
                b.y = B[(k0 + 2 * g + 1) * ldb + (c0 + m)];
            }
            acc = __builtin_amdgcn_wmma_f32_16x16x4_f32(false, a, false, b,
                                                        (short)0, acc, false, false);
        }
#pragma unroll
        for (int r = 0; r < 8; ++r)
            C[(r0 + r + 8 * g) * ldc + (c0 + m)] = acc[r];
    }
}

// ---------- kernel ----------
__global__ __launch_bounds__(256)
void riemann_attn_kernel(const float* __restrict__ x,
                         const float* __restrict__ Wqkv,
                         const float* __restrict__ bqkv,
                         const float* __restrict__ Wfc,
                         const float* __restrict__ bfc,
                         const float* __restrict__ hscale,
                         float* __restrict__ out) {
    __shared__ __align__(16) float Xs[64 * 192];      // 48 KB: x tile, f32
    __shared__ __align__(16) float MB[5 * 64 * 64];   // 80 KB: log/NS scratch
    __shared__ __align__(16) __bf16 QKVh[3 * 64 * 64];// 24 KB: per-head Q,K,V
    __shared__ float red[64];

    float* m0 = MB;             // S -> A0 -> R -> E
    float* m1 = MB + 4096;      // Y / E^2
    float* m2 = MB + 2 * 4096;  // Z / E^3
    float* m3 = MB + 3 * 4096;  // P/T, later SCORES
    float* m4 = MB + 4 * 4096;  // Ynew scratch, later LOGS
    float* SCORES = m3;
    float* LOGS   = m4;
    __bf16* Obf  = (__bf16*)m0;          // 64x192 bf16 over m0 + half m1
    __bf16* ATbf = (__bf16*)(MB + 6144); // 64x64 bf16, second half of m1

    const int tid  = threadIdx.x;
    const int wave = tid >> 5;
    const int lane = tid & 31;
    const int g    = lane >> 4;
    const int m    = lane & 15;
    const int bn   = blockIdx.x;

    // Warm L2 for the (shared, L2-resident) weights
    if (tid < 32) {
        __builtin_prefetch(Wqkv + tid * 2048, 0, 1);
        __builtin_prefetch(Wfc + tid * 1024, 0, 1);
    }

    // ---- load x tile (64 channels x 192 dims) into LDS ----
    const float* xg = x + (size_t)bn * 64 * 192;
#if defined(__gfx1250__) && __has_builtin(__builtin_amdgcn_tensor_load_to_lds)
    if (wave == 0) {
        // Tensor DMA descriptor (D#), ISA ch.8: 2D tile 192x64 f32, row stride 192.
        const unsigned lds_addr = (unsigned)(uintptr_t)(&Xs[0]);   // LDS aperture: low 32 bits = LDS offset
        const unsigned long long ga = (unsigned long long)(uintptr_t)xg;
        u32x4 g0;
        g0[0] = 1u;                                       // count=1, user descriptor
        g0[1] = lds_addr;                                 // lds_addr
        g0[2] = (unsigned)(ga & 0xFFFFFFFFu);             // global_addr[31:0]
        g0[3] = (unsigned)((ga >> 32) & 0x1FFFFFFu) | (2u << 30); // global_addr[56:32], type=2
        i32x8 g1;
        g1[0] = (2 << 16);            // workgroup_mask=0, data_size=2 (4 bytes)
        g1[1] = (192 << 16);          // tensor_dim0[15:0]=192
        g1[2] = (64 << 16);           // tensor_dim0[31:16]=0, tensor_dim1[15:0]=64
        g1[3] = (192 << 16);          // tensor_dim1[31:16]=0, tile_dim0=192
        g1[4] = 64;                   // tile_dim1=64, tile_dim2=0
        g1[5] = 192;                  // tensor_dim0_stride[31:0]=192
        g1[6] = 0;                    // stride[47:32]=0, tensor_dim1_stride[15:0]=0
        g1[7] = 0;
        i32x4 z4 = {0, 0, 0, 0};      // groups 2/3 unused (2D tensor)
        i32x8 z8 = {0, 0, 0, 0, 0, 0, 0, 0};
        // 6-arg form (clang-23 / therock-10.0 toolchain)
        __builtin_amdgcn_tensor_load_to_lds(g0, g1, z4, z4, z8, 0);
    }
    __builtin_amdgcn_s_wait_tensorcnt(0);
#else
    for (int i = tid; i < 3072; i += 256)
        ((float4*)Xs)[i] = ((const float4*)xg)[i];
#endif
    __syncthreads();

    // ---- S = X X^T / D + eps I  (f32 WMMA, K=192) ----
    mm64_f32(Xs, 192, Xs, 192, /*BT=*/true, m0, 64, 192, wave, g, m);
    __syncthreads();
    const float invD = 1.0f / 192.0f;
    for (int i = tid; i < 4096; i += 256) {
        float v = m0[i] * invD;
        if ((i >> 6) == (i & 63)) v += EPSV;
        m0[i] = v;
    }
    __syncthreads();

    // ---- trace normalization: A0 = S / a,  a = tr(S)/64 ----
    if (tid < 64) red[tid] = m0[tid * 65];
    __syncthreads();
    if (tid == 0) { float s = 0.f; for (int i = 0; i < 64; ++i) s += red[i]; red[0] = fmaxf(s * (1.0f / 64.0f), EPSV); }
    __syncthreads();
    const float aN = red[0];
    const float loga = __logf(aN);
    __syncthreads();
    {
        const float inva = 1.0f / aN;
        for (int i = tid; i < 4096; i += 256) m0[i] *= inva;
    }
    __syncthreads();

    // ---- inverse scaling & squaring: 3 Newton-Schulz square roots ----
#pragma unroll 1
    for (int j = 0; j < 3; ++j) {
        if (tid < 64) red[tid] = m0[tid * 65];
        __syncthreads();
        if (tid == 0) { float s = 0.f; for (int i = 0; i < 64; ++i) s += red[i]; red[0] = fmaxf(s * (1.0f / 64.0f), EPSV); }
        __syncthreads();
        const float c = red[0];
        __syncthreads();
        const float invc = 1.0f / c;
        for (int i = tid; i < 4096; i += 256) {
            m1[i] = m0[i] * invc;                         // Y = R/c
            m2[i] = ((i >> 6) == (i & 63)) ? 1.0f : 0.0f; // Z = I
        }
        __syncthreads();
#pragma unroll 1
        for (int it = 0; it < 4; ++it) {
            mm64_f32(m2, 64, m1, 64, false, m3, 64, 64, wave, g, m);   // P = Z@Y
            __syncthreads();
            for (int i = tid; i < 4096; i += 256)                      // T = (3I-P)/2
                m3[i] = (((i >> 6) == (i & 63)) ? 1.5f : 0.0f) - 0.5f * m3[i];
            __syncthreads();
            mm64_f32(m1, 64, m3, 64, false, m4, 64, 64, wave, g, m);   // Ynew = Y@T
            mm64_f32(m3, 64, m2, 64, false, m0, 64, 64, wave, g, m);   // Znew = T@Z
            __syncthreads();
            for (int i = tid; i < 4096; i += 256) { m1[i] = m4[i]; m2[i] = m0[i]; }
            __syncthreads();
        }
        const float sc = __fsqrt_rn(c);
        for (int i = tid; i < 4096; i += 256) m0[i] = m1[i] * sc;      // R = sqrt(c)*Y
        __syncthreads();
    }

    // ---- log via Mercator: E = R - I ; log(S) = 8*(E - E^2/2 + E^3/3) + log(a) I ----
    for (int i = tid; i < 4096; i += 256)
        if ((i >> 6) == (i & 63)) m0[i] -= 1.0f;
    __syncthreads();
    mm64_f32(m0, 64, m0, 64, false, m1, 64, 64, wave, g, m);  // E^2
    __syncthreads();
    mm64_f32(m0, 64, m1, 64, false, m2, 64, 64, wave, g, m);  // E^3
    __syncthreads();
    for (int i = tid; i < 4096; i += 256) {
        float v = 8.0f * (m0[i] - 0.5f * m1[i] + (1.0f / 3.0f) * m2[i]);
        if ((i >> 6) == (i & 63)) v += loga;
        LOGS[i] = v;
    }
    __syncthreads();
    // From here m0/m1 region is reused as Obf / ATbf (bf16 overlays).

    // ---- per-head attention (bf16 WMMA) ----
#pragma unroll 1
    for (int h = 0; h < 3; ++h) {
        const float hsv = hscale[h];
#pragma unroll 1
        for (int s = 0; s < 3; ++s) {
            const int colOff = s * 192 + h * 64;
            __bf16* dst = QKVh + s * 4096;
            for (int t = wave; t < 16; t += 8) {
                const int r0 = (t >> 2) << 4, c0 = (t & 3) << 4;
                v8f acc = {0.f, 0.f, 0.f, 0.f, 0.f, 0.f, 0.f, 0.f};
                for (int k0 = 0; k0 < 192; k0 += 32) {
                    v16bf a = ldA_f32(Xs, 192, r0, k0, g, m);
                    v16bf b = ldB_f32(Wqkv + colOff, 576, k0, c0, g, m);
                    acc = __builtin_amdgcn_wmma_f32_16x16x32_bf16(false, a, false, b,
                                                                  (short)0, acc, false, false);
                }
                const float bias = bqkv[colOff + c0 + m];
#pragma unroll
                for (int r = 0; r < 8; ++r)
                    dst[(r0 + r + 8 * g) * 64 + c0 + m] = f2bf(acc[r] + bias);
            }
        }
        __syncthreads();

        const __bf16* Qh = QKVh;
        const __bf16* Kh = QKVh + 4096;
        const __bf16* Vh = QKVh + 2 * 4096;

        // scores = Q K^T / sqrt(64) + hs * logS
        for (int t = wave; t < 16; t += 8) {
            const int r0 = (t >> 2) << 4, c0 = (t & 3) << 4;
            v8f acc = {0.f, 0.f, 0.f, 0.f, 0.f, 0.f, 0.f, 0.f};
            for (int k0 = 0; k0 < 64; k0 += 32) {
                v16bf a = ldA_bf16(Qh, 64, r0, k0, g, m);
                v16bf b = ldBT_bf16(Kh, 64, k0, c0, g, m);
                acc = __builtin_amdgcn_wmma_f32_16x16x32_bf16(false, a, false, b,
                                                              (short)0, acc, false, false);
            }
#pragma unroll
            for (int r = 0; r < 8; ++r) {
                const int rr = r0 + r + 8 * g, cc = c0 + m;
                SCORES[rr * 64 + cc] = acc[r] * 0.125f + hsv * LOGS[rr * 64 + cc];
            }
        }
        __syncthreads();

        // softmax: 4 lanes per row, shfl_xor reduce across the quad;
        // fused bf16 conversion of attention probs into ATbf.
        {
            const int r = tid >> 2, q = tid & 3;
            float* row = SCORES + r * 64 + q * 16;
            __bf16* arow = ATbf + r * 64 + q * 16;
            float mx = row[0];
#pragma unroll
            for (int i = 1; i < 16; ++i) mx = fmaxf(mx, row[i]);
            mx = fmaxf(mx, __shfl_xor(mx, 1));
            mx = fmaxf(mx, __shfl_xor(mx, 2));
            float s = 0.f;
            float e[16];
#pragma unroll
            for (int i = 0; i < 16; ++i) { e[i] = __expf(row[i] - mx); s += e[i]; }
            s += __shfl_xor(s, 1);
            s += __shfl_xor(s, 2);
            const float inv = 1.0f / s;
#pragma unroll
            for (int i = 0; i < 16; ++i) arow[i] = f2bf(e[i] * inv);
        }
        __syncthreads();

        // out_h = attn @ V -> Obf[:, h*64 .. h*64+64)
        for (int t = wave; t < 16; t += 8) {
            const int r0 = (t >> 2) << 4, c0 = (t & 3) << 4;
            v8f acc = {0.f, 0.f, 0.f, 0.f, 0.f, 0.f, 0.f, 0.f};
            for (int k0 = 0; k0 < 64; k0 += 32) {
                v16bf a = ldA_bf16(ATbf, 64, r0, k0, g, m);
                v16bf b = ldB_bf16(Vh, 64, k0, c0, g, m);
                acc = __builtin_amdgcn_wmma_f32_16x16x32_bf16(false, a, false, b,
                                                              (short)0, acc, false, false);
            }
#pragma unroll
            for (int r = 0; r < 8; ++r)
                Obf[(r0 + r + 8 * g) * 192 + h * 64 + c0 + m] = f2bf(acc[r]);
        }
        __syncthreads();
    }

    // ---- final projection: out = Obf @ Wfc + bfc  (64x192 @ 192x192) ----
    for (int t = wave; t < 48; t += 8) {
        const int r0 = (t / 12) << 4, c0 = (t % 12) << 4;
        v8f acc = {0.f, 0.f, 0.f, 0.f, 0.f, 0.f, 0.f, 0.f};
        for (int k0 = 0; k0 < 192; k0 += 32) {
            v16bf a = ldA_bf16(Obf, 192, r0, k0, g, m);
            v16bf b = ldB_f32(Wfc, 192, k0, c0, g, m);
            acc = __builtin_amdgcn_wmma_f32_16x16x32_bf16(false, a, false, b,
                                                          (short)0, acc, false, false);
        }
#pragma unroll
        for (int r = 0; r < 8; ++r) {
            const int rr = r0 + r + 8 * g, cc = c0 + m;
            out[((size_t)bn * 64 + rr) * 192 + cc] = acc[r] + bfc[cc];
        }
    }
}

extern "C" void kernel_launch(void* const* d_in, const int* in_sizes, int n_in,
                              void* d_out, int out_size, void* d_ws, size_t ws_size,
                              hipStream_t stream) {
    (void)in_sizes; (void)n_in; (void)out_size; (void)d_ws; (void)ws_size;
    const float* x      = (const float*)d_in[0];
    const float* Wqkv   = (const float*)d_in[1];
    const float* bqkv   = (const float*)d_in[2];
    const float* Wfc    = (const float*)d_in[3];
    const float* bfc    = (const float*)d_in[4];
    const float* hscale = (const float*)d_in[5];
    float* out = (float*)d_out;

    dim3 grid(1024);   // B*N = 16*64 independent channel-attention problems
    dim3 block(256);   // 8 wave32 per workgroup
    riemann_attn_kernel<<<grid, block, 0, stream>>>(x, Wqkv, bqkv, Wfc, bfc, hscale, out);
}
